// ESMM_37409165148970
// MI455X (gfx1250) — matrix operand order repaired and checked by reading, not compiled, not run on hardware
//
#include <hip/hip_runtime.h>
#include <hip/hip_bf16.h>
#include <math.h>

typedef __attribute__((ext_vector_type(16))) _Float16 v16h;
typedef __attribute__((ext_vector_type(8)))  _Float16 v8h;
typedef __attribute__((ext_vector_type(8)))  float    v8f;

#define T_  200
#define B_  2048
#define D_  256
#define H1_ 512
#define H2_ 256
#define LL_ 10
#define M_  (B_ * LL_)   // 20480 rows

union AFrag { v16h v; v8h h[2]; };

// ---------------------------------------------------------------------------
// Convert fp32 weight [K,N] -> f16 transposed [N,K] so WMMA B fragments are
// two 8-half b128 loads along k (16 halves apart) per lane.
// ---------------------------------------------------------------------------
__global__ void wcvt_kernel(const float* __restrict__ W, _Float16* __restrict__ Wt,
                            int K, int N) {
  int id = blockIdx.x * blockDim.x + threadIdx.x;
  if (id >= K * N) return;
  int k = id / N, n = id % N;
  Wt[(size_t)n * K + k] = (_Float16)W[(size_t)k * N + n];
}

// ---------------------------------------------------------------------------
// Running prefix sum over time; snapshot the last LL_ steps as prefix means,
// written directly as the f16 WMMA A matrix [M_, D_] row-major.
// One block per batch element, one thread per channel (coalesced over D).
// ---------------------------------------------------------------------------
__global__ __launch_bounds__(256) void prefix_kernel(const float* __restrict__ x,
                                                     const int* __restrict__ lengths,
                                                     _Float16* __restrict__ hA) {
  const int j = blockIdx.x;
  const int d = threadIdx.x;
  const int len = lengths[j];
  const int start = len - LL_;
  const float* p = x + (size_t)j * D_ + d;   // element (t, j, d) at t*B*D
  float s = 0.f;
  for (int t = 0; t < len; ++t) {
    s += p[(size_t)t * (B_ * D_)];
    if (t >= start) {
      const int row = j * LL_ + (t - start);
      hA[(size_t)row * D_ + d] = (_Float16)(s / (float)(t + 1));
    }
  }
}

// ---------------------------------------------------------------------------
// One tower of the fused MLP on a 16-row tile. Pointers come straight from
// kernel args so clang can infer global address space (global_load_b128
// instead of flat_load_b128).
// ---------------------------------------------------------------------------
__device__ __forceinline__ void tower_16rows(
    const _Float16* __restrict__ hA,
    const _Float16* __restrict__ W1t, const float* __restrict__ b1,
    const _Float16* __restrict__ W2t, const float* __restrict__ b2,
    const float* __restrict__ W3, float b3,
    _Float16* y1, float* y2, float* y3row,
    int wave, int lane, int lhalf, int hi, int mbase) {
  // ---------------- GEMM1: h[16,256] x W1[256,512] ----------------
  v8f acc[4] = {};
  {
    const _Float16* aptr = hA + (size_t)(mbase + lhalf) * D_;
#pragma unroll
    for (int ks = 0; ks < D_ / 32; ++ks) {
      const int akb = ks * 32 + hi * 8;
      AFrag a;
      a.h[0] = *(const v8h*)(aptr + akb);        // K = akb .. akb+7
      a.h[1] = *(const v8h*)(aptr + akb + 16);   // K = akb+16 .. akb+23
      const int bkb = ks * 32 + hi * 8;
#pragma unroll
      for (int i = 0; i < 4; ++i) {
        const int ncol = (wave * 4 + i) * 16 + lhalf;
        const _Float16* bp = W1t + (size_t)ncol * D_ + bkb;
        AFrag b;
        b.h[0] = *(const v8h*)(bp);
        b.h[1] = *(const v8h*)(bp + 16);
        acc[i] = __builtin_amdgcn_wmma_f32_16x16x32_f16(
            false, a.v, false, b.v, (short)0, acc[i], false, false);
      }
    }
  }
#pragma unroll
  for (int i = 0; i < 4; ++i) {
    const int ncol = (wave * 4 + i) * 16 + lhalf;
    const float bb = b1[ncol];
#pragma unroll
    for (int v = 0; v < 8; ++v) {
      const int rowl = v + hi * 8;
      float xv = acc[i][v] + bb;
      xv = xv > 0.f ? xv : 0.01f * xv;           // LeakyReLU
      y1[rowl * H1_ + ncol] = (_Float16)xv;
    }
  }
  __syncthreads();

  // ---------------- GEMM2: y1[16,512] x W2[512,256] ----------------
  v8f acc2[2] = {};
#pragma unroll
  for (int ks = 0; ks < H1_ / 32; ++ks) {
    const int akb = ks * 32 + hi * 8;
    AFrag a;
    a.h[0] = *(const v8h*)&y1[lhalf * H1_ + akb];
    a.h[1] = *(const v8h*)&y1[lhalf * H1_ + akb + 16];
    const int bkb = ks * 32 + hi * 8;
#pragma unroll
    for (int i = 0; i < 2; ++i) {
      const int ncol = (wave * 2 + i) * 16 + lhalf;
      const _Float16* bp = W2t + (size_t)ncol * H1_ + bkb;
      AFrag b;
      b.h[0] = *(const v8h*)(bp);
      b.h[1] = *(const v8h*)(bp + 16);
      acc2[i] = __builtin_amdgcn_wmma_f32_16x16x32_f16(
          false, a.v, false, b.v, (short)0, acc2[i], false, false);
    }
  }
#pragma unroll
  for (int i = 0; i < 2; ++i) {
    const int ncol = (wave * 2 + i) * 16 + lhalf;
    const float bb = b2[ncol];
#pragma unroll
    for (int v = 0; v < 8; ++v) {
      const int rowl = v + hi * 8;
      float xv = acc2[i][v] + bb;
      y2[rowl * H2_ + ncol] = xv > 0.f ? xv : 0.01f * xv;
    }
  }
  __syncthreads();

  // ---------------- Layer 3: per-row 256-dot (VALU + wave reduce) ----
#pragma unroll
  for (int rr = 0; rr < 2; ++rr) {
    const int r = wave * 2 + rr;
    float p = 0.f;
#pragma unroll
    for (int d = 0; d < H2_ / 32; ++d) p += y2[r * H2_ + d * 32 + lane] * W3[d * 32 + lane];
#pragma unroll
    for (int off = 16; off > 0; off >>= 1) p += __shfl_xor(p, off, 32);
    if (lane == 0) {
      float xv = p + b3;
      y3row[r] = xv > 0.f ? xv : 0.01f * xv;
    }
  }
  __syncthreads();
}

// ---------------------------------------------------------------------------
// Fused two-tower MLP on a 16-row tile. 8 wave32 per block.
//  GEMM1: [16,256]x[256,512]  (8 k-steps, 32 n-tiles -> 4 per wave)
//  GEMM2: [16,512]x[512,256]  (16 k-steps, 16 n-tiles -> 2 per wave)
//  L3: per-row 256-dot + LeakyReLU; sigmoid(ctr)*sigmoid(cvr) -> out
// ---------------------------------------------------------------------------
__global__ __launch_bounds__(256) void mlp_kernel(
    const _Float16* __restrict__ hA,
    const _Float16* __restrict__ W1t_ctr, const float* __restrict__ b1_ctr,
    const _Float16* __restrict__ W2t_ctr, const float* __restrict__ b2_ctr,
    const float* __restrict__ W3_ctr,     const float* __restrict__ b3_ctr,
    const _Float16* __restrict__ W1t_cvr, const float* __restrict__ b1_cvr,
    const _Float16* __restrict__ W2t_cvr, const float* __restrict__ b2_cvr,
    const float* __restrict__ W3_cvr,     const float* __restrict__ b3_cvr,
    float* __restrict__ out) {
  __shared__ __align__(16) _Float16 y1[16 * H1_];   // 16 KB
  __shared__ __align__(16) float    y2[16 * H2_];   // 16 KB
  __shared__ float y3[2][16];

  const int wave  = threadIdx.x >> 5;
  const int lane  = threadIdx.x & 31;
  const int lhalf = lane & 15;         // column (B/C/D) or row-in-tile (A)
  const int hi    = (lane >> 4) & 1;   // lane-group select
  const int mbase = blockIdx.x * 16;

  tower_16rows(hA, W1t_ctr, b1_ctr, W2t_ctr, b2_ctr, W3_ctr, b3_ctr[0],
               y1, y2, y3[0], wave, lane, lhalf, hi, mbase);
  tower_16rows(hA, W1t_cvr, b1_cvr, W2t_cvr, b2_cvr, W3_cvr, b3_cvr[0],
               y1, y2, y3[1], wave, lane, lhalf, hi, mbase);

  // sigmoid(ctr) * sigmoid(cvr)
  if (threadIdx.x < 16) {
    const float a = y3[0][threadIdx.x];
    const float b = y3[1][threadIdx.x];
    const float sa = 1.f / (1.f + __expf(-a));
    const float sb = 1.f / (1.f + __expf(-b));
    out[mbase + threadIdx.x] = sa * sb;
  }
}

// ---------------------------------------------------------------------------
extern "C" void kernel_launch(void* const* d_in, const int* in_sizes, int n_in,
                              void* d_out, int out_size, void* d_ws, size_t ws_size,
                              hipStream_t stream) {
  (void)in_sizes; (void)n_in; (void)out_size; (void)ws_size;

  const float* inputs  = (const float*)d_in[0];
  const int*   lengths = (const int*)d_in[1];
  // d_in[2] = label_len (compile-time constant 10 in reference)
  const float* ctr_W1 = (const float*)d_in[3];
  const float* ctr_b1 = (const float*)d_in[4];
  const float* ctr_W2 = (const float*)d_in[5];
  const float* ctr_b2 = (const float*)d_in[6];
  const float* ctr_W3 = (const float*)d_in[7];
  const float* ctr_b3 = (const float*)d_in[8];
  const float* cvr_W1 = (const float*)d_in[9];
  const float* cvr_b1 = (const float*)d_in[10];
  const float* cvr_W2 = (const float*)d_in[11];
  const float* cvr_b2 = (const float*)d_in[12];
  const float* cvr_W3 = (const float*)d_in[13];
  const float* cvr_b3 = (const float*)d_in[14];

  // Workspace carve-out (~11.5 MB total)
  char* ws = (char*)d_ws;
  size_t off = 0;
  _Float16* hA      = (_Float16*)(ws + off); off += (size_t)M_ * D_ * sizeof(_Float16);
  _Float16* W1t_ctr = (_Float16*)(ws + off); off += (size_t)D_  * H1_ * sizeof(_Float16);
  _Float16* W2t_ctr = (_Float16*)(ws + off); off += (size_t)H1_ * H2_ * sizeof(_Float16);
  _Float16* W1t_cvr = (_Float16*)(ws + off); off += (size_t)D_  * H1_ * sizeof(_Float16);
  _Float16* W2t_cvr = (_Float16*)(ws + off); off += (size_t)H1_ * H2_ * sizeof(_Float16);

  // Weight convert + transpose (f32 [K,N] -> f16 [N,K])
  wcvt_kernel<<<(D_ * H1_ + 255) / 256, 256, 0, stream>>>(ctr_W1, W1t_ctr, D_, H1_);
  wcvt_kernel<<<(H1_ * H2_ + 255) / 256, 256, 0, stream>>>(ctr_W2, W2t_ctr, H1_, H2_);
  wcvt_kernel<<<(D_ * H1_ + 255) / 256, 256, 0, stream>>>(cvr_W1, W1t_cvr, D_, H1_);
  wcvt_kernel<<<(H1_ * H2_ + 255) / 256, 256, 0, stream>>>(cvr_W2, W2t_cvr, H1_, H2_);

  // Prefix means -> f16 A matrix
  prefix_kernel<<<B_, 256, 0, stream>>>(inputs, lengths, hA);

  // Fused two-tower MLP (WMMA) -> out [B*10]
  mlp_kernel<<<M_ / 16, 256, 0, stream>>>(
      hA,
      W1t_ctr, ctr_b1, W2t_ctr, ctr_b2, ctr_W3, ctr_b3,
      W1t_cvr, cvr_b1, W2t_cvr, cvr_b2, cvr_W3, cvr_b3,
      (float*)d_out);
}
